// KANLayer_2989297238631
// MI455X (gfx1250) — compile-verified, matrix operand order checked by reading, not run on hardware
//
#include <hip/hip_runtime.h>
#include <hip/hip_bf16.h>

// ---------------------------------------------------------------------------
// KAN layer as one bf16 WMMA GEMM on gfx1250 (MI455X, wave32).
//   out = selu( A[4096 x 4096] @ Bt^T[4096 x 1024] + cbias )
//   A[b, 4i+d]  = {x, tanh(x), tanh^2, tanh^3}
//   Bt[j, 4i+d] = {W[i,j], g*s1, g*s2, g*s3}   (d=0 spline term folded into cbias)
// GEMM inner loop: double-buffered LDS fed by gfx1250 async global->LDS DMA
// (ASYNCcnt / s_wait_asynccnt), 8x v_wmma_f32_16x16x32_bf16 per K-step.
// ---------------------------------------------------------------------------

typedef __attribute__((ext_vector_type(16))) __bf16 v16bf;
typedef __attribute__((ext_vector_type(8)))  float  v8f;
typedef __attribute__((ext_vector_type(4)))  int    v4i;

#define BDIM 4096
#define IDIM 1024
#define UDIM 1024
#define KDIM 4096   // 4 * IDIM

#define TM 128
#define TN 128
#define TK 32
#define NTILES (KDIM / TK)
#define LPAD (TK + 8)   // 40 halves = 80B row stride, 16B aligned

#if defined(__has_builtin)
#if __has_builtin(__builtin_amdgcn_global_load_async_to_lds_b128) && \
    __has_builtin(__builtin_amdgcn_s_wait_asynccnt)
#define USE_ASYNC_LDS 1
#endif
#endif

#define AS1 __attribute__((address_space(1)))
#define AS3 __attribute__((address_space(3)))

// float -> bf16 bit pattern, round-to-nearest-even
__device__ __forceinline__ unsigned short f2bf(float f) {
    unsigned int u = __float_as_uint(f);
    unsigned int r = (u + 0x7FFFu + ((u >> 16) & 1u)) >> 16;
    return (unsigned short)r;
}

__device__ __forceinline__ float selu_f(float x) {
    const float scale = 1.0507009873554805f;
    const float alpha = 1.6732632423543772f;
    return x > 0.f ? scale * x : scale * alpha * (__expf(x) - 1.f);
}

// ---- Prep A: basis [x, t, t^2, t^3] packed bf16, A[b, 4i+d] ----------------
__global__ __launch_bounds__(256) void kan_prep_a(const float* __restrict__ x,
                                                  unsigned short* __restrict__ A) {
    int idx = blockIdx.x * 256 + threadIdx.x;      // over B*I
    float v  = x[idx];
    float t  = tanhf(v);
    float t2 = t * t;
    uint2 p;
    p.x = (unsigned)f2bf(v)  | ((unsigned)f2bf(t)      << 16);
    p.y = (unsigned)f2bf(t2) | ((unsigned)f2bf(t2 * t) << 16);
    *reinterpret_cast<uint2*>(A + (size_t)4 * idx) = p;
}

// ---- Prep B: Bt[j, 4i+d] = {W, g*s1, g*s2, g*s3} ---------------------------
__global__ __launch_bounds__(256) void kan_prep_b(const float* __restrict__ W,
                                                  const float* __restrict__ spl,
                                                  const float* __restrict__ gate,
                                                  unsigned short* __restrict__ Bt) {
    int idx = blockIdx.x * 256 + threadIdx.x;      // over I*U, i fastest
    int i = idx & (IDIM - 1);
    int j = idx >> 10;
    size_t ij = (size_t)i * UDIM + j;
    float w = W[ij];
    float g = gate[ij];
    const float* s = spl + ij * 4;
    uint2 p;
    p.x = (unsigned)f2bf(w)        | ((unsigned)f2bf(g * s[1]) << 16);
    p.y = (unsigned)f2bf(g * s[2]) | ((unsigned)f2bf(g * s[3]) << 16);
    *reinterpret_cast<uint2*>(Bt + (size_t)j * KDIM + (size_t)4 * i) = p;
}

// ---- Prep bias: cbias[j] = bias[j] + sum_i gate[i,j]*spline[i,j,0] ---------
__global__ __launch_bounds__(256) void kan_prep_bias(const float* __restrict__ bias,
                                                     const float* __restrict__ spl,
                                                     const float* __restrict__ gate,
                                                     float* __restrict__ cbias) {
    __shared__ float red[256];
    int j = blockIdx.x;
    float s = 0.f;
    for (int i = threadIdx.x; i < IDIM; i += 256) {
        size_t ij = (size_t)i * UDIM + j;
        s += gate[ij] * spl[ij * 4];
    }
    red[threadIdx.x] = s;
    __syncthreads();
    for (int off = 128; off > 0; off >>= 1) {
        if (threadIdx.x < (unsigned)off) red[threadIdx.x] += red[threadIdx.x + off];
        __syncthreads();
    }
    if (threadIdx.x == 0) cbias[j] = red[0] + bias[j];
}

// ---- Main GEMM -------------------------------------------------------------
union Frag {
    v16bf v;
    uint4 q[2];
};

#if USE_ASYNC_LDS
// One 16-byte async DMA: global -> LDS, no VGPR round trip (ASYNCcnt tracked).
// Builtin signature (from hipcc sema): (v4i AS1* gsrc, v4i AS3* ldst, imm, imm).
__device__ __forceinline__ void async_cp16(const unsigned short* g, unsigned short* l) {
    __builtin_amdgcn_global_load_async_to_lds_b128((AS1 v4i*)g, (AS3 v4i*)l, 0, 0);
}

// Each thread issues 4 b128 DMAs: 2 for the A tile, 2 for the Bt tile.
__device__ __forceinline__ void issue_tile(const unsigned short* __restrict__ A,
                                           const unsigned short* __restrict__ Bt,
                                           unsigned short (*la)[LPAD],
                                           unsigned short (*lb)[LPAD],
                                           int rowBase, int colBase, int kt, int tid) {
#pragma unroll
    for (int it = 0; it < 2; ++it) {
        int item = tid + it * 256;
        int r  = item >> 2;
        int qc = item & 3;
        async_cp16(A  + (size_t)(rowBase + r) * KDIM + kt + qc * 8, &la[r][qc * 8]);
        async_cp16(Bt + (size_t)(colBase + r) * KDIM + kt + qc * 8, &lb[r][qc * 8]);
    }
}
#endif

__global__ __launch_bounds__(256) void kan_gemm(const unsigned short* __restrict__ A,
                                                const unsigned short* __restrict__ Bt,
                                                const float* __restrict__ cbias,
                                                float* __restrict__ out) {
    const int tid  = threadIdx.x;
    const int lane = tid & 31;
    const int wave = tid >> 5;        // 0..7
    const int wm   = wave & 1;        // M: 2 waves * 64 rows
    const int wn   = wave >> 1;       // N: 4 waves * 32 cols
    const int m16  = lane & 15;
    const int h    = lane >> 4;       // lane half (K-split per ISA 16-bit layout)

    const int rowBase = blockIdx.y * TM;
    const int colBase = blockIdx.x * TN;

    v8f acc[4][2];
    const v8f vzero = {0.f, 0.f, 0.f, 0.f, 0.f, 0.f, 0.f, 0.f};
#pragma unroll
    for (int mi = 0; mi < 4; ++mi)
#pragma unroll
        for (int ni = 0; ni < 2; ++ni) acc[mi][ni] = vzero;

#if USE_ASYNC_LDS
    // Double-buffered LDS tiles (2 x ~20KB), filled by async DMA.
    __shared__ __align__(16) unsigned short lA[2][TM][LPAD];
    __shared__ __align__(16) unsigned short lB[2][TN][LPAD];

    issue_tile(A, Bt, lA[0], lB[0], rowBase, colBase, 0, tid);

    for (int s = 0; s < NTILES; ++s) {
        const int cur = s & 1;
        if (s + 1 < NTILES) {
            // Prefetch next tile into the other buffer while this one computes.
            issue_tile(A, Bt, lA[cur ^ 1], lB[cur ^ 1], rowBase, colBase,
                       (s + 1) * TK, tid);
            __builtin_amdgcn_s_wait_asynccnt(4);   // tile s landed; s+1 in flight
        } else {
            __builtin_amdgcn_s_wait_asynccnt(0);   // final tile landed
        }
        __syncthreads();   // all threads' DMAs for tile s visible

        Frag fb[2];
#pragma unroll
        for (int ni = 0; ni < 2; ++ni) {
            int n = wn * 32 + ni * 16 + m16;
            fb[ni].q[0] = *reinterpret_cast<const uint4*>(&lB[cur][n][8 * h]);
            fb[ni].q[1] = *reinterpret_cast<const uint4*>(&lB[cur][n][16 + 8 * h]);
        }
#pragma unroll
        for (int mi = 0; mi < 4; ++mi) {
            int m = wm * 64 + mi * 16 + m16;
            Frag fa;
            fa.q[0] = *reinterpret_cast<const uint4*>(&lA[cur][m][8 * h]);
            fa.q[1] = *reinterpret_cast<const uint4*>(&lA[cur][m][16 + 8 * h]);
#pragma unroll
            for (int ni = 0; ni < 2; ++ni) {
                acc[mi][ni] = __builtin_amdgcn_wmma_f32_16x16x32_bf16(
                    false, fa.v, false, fb[ni].v, (short)0, acc[mi][ni], false, false);
            }
        }
        __syncthreads();   // buffer cur free for the DMA issued at iteration s+1
    }
#else
    // Fallback: single-buffer, register-batched staging (no async builtins).
    __shared__ __align__(16) unsigned short lA1[TM][LPAD];
    __shared__ __align__(16) unsigned short lB1[TN][LPAD];

    for (int kt = 0; kt < KDIM; kt += TK) {
        uint4 ra[2], rb[2];
#pragma unroll
        for (int it = 0; it < 2; ++it) {
            int item = tid + it * 256;
            int r  = item >> 2;
            int qc = item & 3;
            ra[it] = *reinterpret_cast<const uint4*>(A  + (size_t)(rowBase + r) * KDIM + kt + qc * 8);
            rb[it] = *reinterpret_cast<const uint4*>(Bt + (size_t)(colBase + r) * KDIM + kt + qc * 8);
        }
        __syncthreads();
#pragma unroll
        for (int it = 0; it < 2; ++it) {
            int item = tid + it * 256;
            int r  = item >> 2;
            int qc = item & 3;
            *reinterpret_cast<uint4*>(&lA1[r][qc * 8]) = ra[it];
            *reinterpret_cast<uint4*>(&lB1[r][qc * 8]) = rb[it];
        }
        __syncthreads();

        Frag fb[2];
#pragma unroll
        for (int ni = 0; ni < 2; ++ni) {
            int n = wn * 32 + ni * 16 + m16;
            fb[ni].q[0] = *reinterpret_cast<const uint4*>(&lB1[n][8 * h]);
            fb[ni].q[1] = *reinterpret_cast<const uint4*>(&lB1[n][16 + 8 * h]);
        }
#pragma unroll
        for (int mi = 0; mi < 4; ++mi) {
            int m = wm * 64 + mi * 16 + m16;
            Frag fa;
            fa.q[0] = *reinterpret_cast<const uint4*>(&lA1[m][8 * h]);
            fa.q[1] = *reinterpret_cast<const uint4*>(&lA1[m][16 + 8 * h]);
#pragma unroll
            for (int ni = 0; ni < 2; ++ni) {
                acc[mi][ni] = __builtin_amdgcn_wmma_f32_16x16x32_bf16(
                    false, fa.v, false, fb[ni].v, (short)0, acc[mi][ni], false, false);
            }
        }
    }
#endif

    // Epilogue: C/D layout (§7.12.2): VGPR v -> row = v + 8*h, col = lane&15.
#pragma unroll
    for (int ni = 0; ni < 2; ++ni) {
        int col = colBase + wn * 32 + ni * 16 + m16;
        float cb = cbias[col];
#pragma unroll
        for (int mi = 0; mi < 4; ++mi) {
#pragma unroll
            for (int v = 0; v < 8; ++v) {
                int row = rowBase + wm * 64 + mi * 16 + v + 8 * h;
                out[(size_t)row * UDIM + col] = selu_f(acc[mi][ni][v] + cb);
            }
        }
    }
}

// ---------------------------------------------------------------------------
extern "C" void kernel_launch(void* const* d_in, const int* in_sizes, int n_in,
                              void* d_out, int out_size, void* d_ws, size_t ws_size,
                              hipStream_t stream) {
    const float* x    = (const float*)d_in[0];   // [4096,1024]
    const float* W    = (const float*)d_in[1];   // [1024,1024]
    const float* bias = (const float*)d_in[2];   // [1024]
    const float* spl  = (const float*)d_in[3];   // [1024,1024,4]
    const float* gate = (const float*)d_in[4];   // [1024,1024]
    float* out = (float*)d_out;                  // [4096,1024]

    // Workspace layout: A bf16 [4096,4096] | Bt bf16 [1024,4096] | cbias f32 [1024]
    unsigned short* A  = (unsigned short*)d_ws;
    unsigned short* Bt = A + (size_t)BDIM * KDIM;
    float* cbias       = (float*)(Bt + (size_t)UDIM * KDIM);

    kan_prep_a<<<(BDIM * IDIM) / 256, 256, 0, stream>>>(x, A);
    kan_prep_b<<<(IDIM * UDIM) / 256, 256, 0, stream>>>(W, spl, gate, Bt);
    kan_prep_bias<<<UDIM, 256, 0, stream>>>(bias, spl, gate, cbias);

    dim3 grid(UDIM / TN, BDIM / TM);   // 8 x 32 = 256 workgroups
    kan_gemm<<<grid, 256, 0, stream>>>(A, Bt, cbias, out);
}
